// RotaryCascadeAttention_74577812127922
// MI455X (gfx1250) — matrix operand order repaired and checked by reading, not compiled, not run on hardware
//
#include <hip/hip_runtime.h>

typedef _Float16 h8  __attribute__((ext_vector_type(8)));
typedef _Float16 h16 __attribute__((ext_vector_type(16)));
typedef float    v8f __attribute__((ext_vector_type(8)));

union H16 { h16 v; h8 h[2]; _Float16 e[16]; };

// A-fragment (16x32 f16): lane<16 -> row=lane, K {0..7,16..23}; lane>=16 -> row=lane-16, K {8..15,24..31}
__device__ inline h16 load_a_frag(const _Float16* __restrict__ base, int rs, int lane) {
    int row = lane & 15;
    int kb  = (lane < 16) ? 0 : 8;
    const _Float16* p = base + row * rs + kb;
    H16 a;
    a.h[0] = *reinterpret_cast<const h8*>(p);
    a.h[1] = *reinterpret_cast<const h8*>(p + 16);
    return a.v;
}

// B-fragment (32x16 f16) from Bt[N][K] row-major: lane<16 -> col=lane, K {0..7,16..23}; lane>=16 -> col=lane-16, K {8..15,24..31}
__device__ inline h16 load_b_frag(const _Float16* __restrict__ base, int rs, int lane) {
    int col = lane & 15;
    int kb  = (lane < 16) ? 0 : 8;
    const _Float16* p = base + col * rs + kb;
    H16 b;
    b.h[0] = *reinterpret_cast<const h8*>(p);
    b.h[1] = *reinterpret_cast<const h8*>(p + 16);
    return b.v;
}

// ---- CDNA5 async global->LDS copy (ASYNCcnt-tracked) ----
__device__ inline void async_load_b128(unsigned lds_addr, const _Float16* gptr) {
    asm volatile("global_load_async_to_lds_b128 %0, %1, off"
                 :: "v"(lds_addr), "v"(gptr) : "memory");
}
__device__ inline void wait_async0() {
    asm volatile("s_wait_asynccnt 0" ::: "memory");
}
__device__ inline unsigned lds_addr_of(const void* p) {
    // generic LDS address: addr[31:0] is the LDS byte offset (ISA aperture rules)
    return (unsigned)(uintptr_t)p;
}

__global__ void cast_f32_to_f16(const float* __restrict__ src, _Float16* __restrict__ dst, int n) {
    int i = blockIdx.x * blockDim.x + threadIdx.x;
    if (i < n) dst[i] = (_Float16)src[i];
}

// W: [K][N] f32  ->  WT: [N][K] f16
__global__ void transpose_cast(const float* __restrict__ W, _Float16* __restrict__ WT, int K, int N) {
    int i = blockIdx.x * blockDim.x + threadIdx.x;
    if (i < K * N) {
        int n = i / K, k = i - n * K;
        WT[i] = (_Float16)W[(long)k * N + n];
    }
}

// C[M][N] f32 = A[M][K] f16 @ Bt[N][K]^T f16.
// Block tile 32(M) x 256(N), K-step 32. Double-buffered async LDS staging.
__global__ __launch_bounds__(256) void gemm_f16f32(const _Float16* __restrict__ A,
                                                   const _Float16* __restrict__ Bt,
                                                   float* __restrict__ C, int M, int N, int K) {
    __shared__ __attribute__((aligned(16))) _Float16 Abuf[2][32 * 32];    // 2 x 2KB
    __shared__ __attribute__((aligned(16))) _Float16 Bbuf[2][256 * 32];   // 2 x 16KB

    int tid  = threadIdx.x;
    int lane = tid & 31;
    int wave = tid >> 5;
    long arow0 = (long)blockIdx.y * 32;
    long bcol0 = (long)blockIdx.x * 256;

    unsigned aoff[2] = { lds_addr_of(&Abuf[0][0]), lds_addr_of(&Abuf[1][0]) };
    unsigned boff[2] = { lds_addr_of(&Bbuf[0][0]), lds_addr_of(&Bbuf[1][0]) };

    int nsteps = K / 32;
    // issue K-step 0 into buffer 0
    {
        const _Float16* gb = Bt + (bcol0 + tid) * (long)K;
        unsigned lb = boff[0] + tid * 64;
        async_load_b128(lb,      gb);
        async_load_b128(lb + 16, gb + 8);
        async_load_b128(lb + 32, gb + 16);
        async_load_b128(lb + 48, gb + 24);
        if (tid < 128) {
            int r = tid >> 2, o = (tid & 3) * 8;
            async_load_b128(aoff[0] + tid * 16, A + (arow0 + r) * (long)K + o);
        }
    }

    v8f acc[4] = {};
    int mloc = (wave >> 2) * 16;   // local M base: 0 or 16
    int nloc = (wave & 3) * 64;    // local N base

    for (int i = 0; i < nsteps; i++) {
        wait_async0();          // this wave's copies into buf[i&1] done
        __syncthreads();        // => all waves' copies done; prior compute done block-wide
        if (i + 1 < nsteps) {   // issue next K-step into the other buffer
            int k0 = (i + 1) * 32;
            const _Float16* gb = Bt + (bcol0 + tid) * (long)K + k0;
            unsigned lb = boff[(i + 1) & 1] + tid * 64;
            async_load_b128(lb,      gb);
            async_load_b128(lb + 16, gb + 8);
            async_load_b128(lb + 32, gb + 16);
            async_load_b128(lb + 48, gb + 24);
            if (tid < 128) {
                int r = tid >> 2, o = (tid & 3) * 8;
                async_load_b128(aoff[(i + 1) & 1] + tid * 16, A + (arow0 + r) * (long)K + k0 + o);
            }
        }
        const _Float16* Ab = &Abuf[i & 1][0];
        const _Float16* Bb = &Bbuf[i & 1][0];
        // batch ALL fragment ds_loads before the WMMAs -> partial dscnt waits, DS pipe stays ahead
        h16 a  = load_a_frag(Ab + mloc * 32, 32, lane);
        h16 b0 = load_b_frag(Bb + (nloc +  0) * 32, 32, lane);
        h16 b1 = load_b_frag(Bb + (nloc + 16) * 32, 32, lane);
        h16 b2 = load_b_frag(Bb + (nloc + 32) * 32, 32, lane);
        h16 b3 = load_b_frag(Bb + (nloc + 48) * 32, 32, lane);
        acc[0] = __builtin_amdgcn_wmma_f32_16x16x32_f16(false, a, false, b0, (short)0, acc[0], false, false);
        acc[1] = __builtin_amdgcn_wmma_f32_16x16x32_f16(false, a, false, b1, (short)0, acc[1], false, false);
        acc[2] = __builtin_amdgcn_wmma_f32_16x16x32_f16(false, a, false, b2, (short)0, acc[2], false, false);
        acc[3] = __builtin_amdgcn_wmma_f32_16x16x32_f16(false, a, false, b3, (short)0, acc[3], false, false);
        // no trailing barrier needed: next iteration's wait+barrier protects the buffer swap
    }

    int mrow  = (int)arow0 + mloc + ((lane < 16) ? 0 : 8);
    int ncol0 = (int)bcol0 + nloc + (lane & 15);
#pragma unroll
    for (int t = 0; t < 4; t++)
#pragma unroll
        for (int r = 0; r < 8; r++)
            C[(long)(mrow + r) * N + ncol0 + t * 16] = acc[t][r];
}

// qkv[4096][3072] f32 -> q_rot/k_rot f16 [B*H][T][64] with per-head RoPE, vT f16 [B*H][64][T]
__global__ void rope_split_kernel(const float* __restrict__ qkv, _Float16* __restrict__ qr,
                                  _Float16* __restrict__ kr, _Float16* __restrict__ vT) {
    int i = blockIdx.x * blockDim.x + threadIdx.x;  // B*T*H*32 = 2M
    int j = i & 31;
    int h = (i >> 5) & 15;
    int t = (i >> 9) & 1023;
    int b = i >> 19;
    const float* row = qkv + (long)(b * 1024 + t) * 3072;
    // angle = t * 10000^(h/16 - j/32)
    float ang = (float)t * __expf(9.210340372f * ((float)h * (1.0f / 16.0f) - (float)j * (1.0f / 32.0f)));
    float s, c;
    __sincosf(ang, &s, &c);
    int qc = h * 64 + 2 * j;
    float q1 = row[qc],        q2 = row[qc + 1];
    float k1 = row[1024 + qc], k2 = row[1024 + qc + 1];
    long outb = ((long)(b * 16 + h) * 1024 + t) * 64;
    qr[outb + j]      = (_Float16)(q1 * c - q2 * s);
    qr[outb + j + 32] = (_Float16)(q1 * s + q2 * c);
    kr[outb + j]      = (_Float16)(k1 * c - k2 * s);
    kr[outb + j + 32] = (_Float16)(k1 * s + k2 * c);
    long vtb = (long)(b * 16 + h) * 64;
    vT[(vtb + j) * 1024 + t]      = (_Float16)row[2048 + h * 64 + j];
    vT[(vtb + j + 32) * 1024 + t] = (_Float16)row[2048 + h * 64 + j + 32];
}

// One head: grid = B*64 blocks, each handles 16 query rows x all 1024 keys.
__global__ __launch_bounds__(256) void attn_head_kernel(
    const _Float16* __restrict__ qr, const _Float16* __restrict__ kr,
    const _Float16* __restrict__ vT, const float* __restrict__ bias_in,
    const float* __restrict__ cascade_w, _Float16* __restrict__ attout,
    float* __restrict__ bias_part, int head) {
    __shared__ __attribute__((aligned(16))) float sm[16 * 1024];  // 64 KB: logits f32, later probs f16 (aliased)
    _Float16* pf16 = reinterpret_cast<_Float16*>(sm);

    int lane = threadIdx.x & 31;
    int wave = threadIdx.x >> 5;
    int b    = blockIdx.x >> 6;
    int tile = blockIdx.x & 63;
    int q0   = tile * 16;
    int l15  = lane & 15;
    int mbase = (lane < 16) ? 0 : 8;

    const _Float16* qbase = qr + ((long)(b * 16 + head) * 1024 + q0) * 64;
    const _Float16* kbase = kr + (long)(b * 16 + head) * 1024 * 64;
    const _Float16* vbase = vT + (long)(b * 16 + head) * 64 * 1024;

    h16 a0 = load_a_frag(qbase, 64, lane);       // d 0..31
    h16 a1 = load_a_frag(qbase + 32, 64, lane);  // d 32..63

    float cw = (head > 0) ? cascade_w[head - 1] : 0.0f;
    const float scale = 0.125f;                  // 1/sqrt(64)
    const float NEG_INF = -__builtin_huge_valf();

    // Phase 1: logits -> LDS (each wave: 8 key tiles of 16), next-tile K fragments prefetched
    {
        h16 b0 = load_b_frag(kbase + wave * 16 * 64, 64, lane);
        h16 b1 = load_b_frag(kbase + wave * 16 * 64 + 32, 64, lane);
        for (int nt = wave; nt < 64; nt += 8) {
            h16 nb0 = b0, nb1 = b1;
            if (nt + 8 < 64) {
                nb0 = load_b_frag(kbase + (nt + 8) * 16 * 64, 64, lane);
                nb1 = load_b_frag(kbase + (nt + 8) * 16 * 64 + 32, 64, lane);
            }
            v8f acc = {};
            acc = __builtin_amdgcn_wmma_f32_16x16x32_f16(false, a0, false, b0, (short)0, acc, false, false);
            acc = __builtin_amdgcn_wmma_f32_16x16x32_f16(false, a1, false, b1, (short)0, acc, false, false);
            int scol = nt * 16 + l15;
            float biasv = (head > 0) ? cw * bias_in[b * 1024 + scol] : 0.0f;
#pragma unroll
            for (int r = 0; r < 8; r++) {
                int m = mbase + r;
                float v = acc[r] * scale + biasv;
                if (scol > q0 + m) v = NEG_INF;  // causal mask
                sm[m * 1024 + scol] = v;
            }
            b0 = nb0; b1 = nb1;
        }
    }
    __syncthreads();

    // Phase 2: softmax, rows 2*wave and 2*wave+1, kept in registers
    float e0[32], e1[32];
    float inv0, inv1;
    {
        const float* r0 = sm + (2 * wave) * 1024;
        const float* r1 = r0 + 1024;
        float mx0 = NEG_INF, mx1 = NEG_INF;
        for (int u = 0; u < 32; u++) { e0[u] = r0[lane + 32 * u]; mx0 = fmaxf(mx0, e0[u]); }
        for (int u = 0; u < 32; u++) { e1[u] = r1[lane + 32 * u]; mx1 = fmaxf(mx1, e1[u]); }
        for (int off = 16; off > 0; off >>= 1) {
            mx0 = fmaxf(mx0, __shfl_xor(mx0, off, 32));
            mx1 = fmaxf(mx1, __shfl_xor(mx1, off, 32));
        }
        float s0 = 0.f, s1 = 0.f;
        for (int u = 0; u < 32; u++) { e0[u] = __expf(e0[u] - mx0); s0 += e0[u]; }
        for (int u = 0; u < 32; u++) { e1[u] = __expf(e1[u] - mx1); s1 += e1[u]; }
        for (int off = 16; off > 0; off >>= 1) {
            s0 += __shfl_xor(s0, off, 32);
            s1 += __shfl_xor(s1, off, 32);
        }
        inv0 = 1.0f / s0;
        inv1 = 1.0f / s1;
    }
    __syncthreads();  // all f32 logits consumed; safe to alias LDS as f16 probs
    for (int u = 0; u < 32; u++) pf16[(2 * wave) * 1024 + lane + 32 * u]     = (_Float16)(e0[u] * inv0);
    for (int u = 0; u < 32; u++) pf16[(2 * wave + 1) * 1024 + lane + 32 * u] = (_Float16)(e1[u] * inv1);
    __syncthreads();

    if (wave >= 4) {
        // Phase 3: per-block column sums of probs (cascade bias partials), deterministic (no atomics)
        int base = threadIdx.x - 128;
        for (int c = base; c < 1024; c += 128) {
            float ssum = 0.f;
#pragma unroll
            for (int m = 0; m < 16; m++) ssum += (float)pf16[m * 1024 + c];
            bias_part[(long)(b * 64 + tile) * 1024 + c] = ssum;
        }
    } else {
        // Phase 4: out = P @ V, waves 0..3 each own a 16-wide d tile; next V fragment prefetched
        int d0 = wave * 16;
        v8f acc = {};
        int kadd = (lane < 16) ? 0 : 8;
        h16 bf = load_b_frag(vbase + d0 * 1024, 1024, lane);
        for (int kk = 0; kk < 32; kk++) {
            int sb = kk * 32;
            h16 nbf = bf;
            if (kk + 1 < 32) nbf = load_b_frag(vbase + d0 * 1024 + sb + 32, 1024, lane);
            H16 af;
            const _Float16* pr = pf16 + l15 * 1024 + sb + kadd;
            af.h[0] = *reinterpret_cast<const h8*>(pr);
            af.h[1] = *reinterpret_cast<const h8*>(pr + 16);
            acc = __builtin_amdgcn_wmma_f32_16x16x32_f16(false, af.v, false, bf, (short)0, acc, false, false);
            bf = nbf;
        }
        long orow0 = (long)(b * 1024 + q0 + mbase) * 1024 + head * 64 + d0 + l15;
#pragma unroll
        for (int r = 0; r < 8; r++)
            attout[orow0 + (long)r * 1024] = (_Float16)acc[r];
    }
}

__global__ void reduce_bias_kernel(const float* __restrict__ bias_part, float* __restrict__ bias) {
    int i = blockIdx.x * blockDim.x + threadIdx.x;  // 4096 = B*T
    int b = i >> 10, s = i & 1023;
    float sum = 0.f;
    for (int k = 0; k < 64; k++) sum += bias_part[(long)(b * 64 + k) * 1024 + s];
    bias[i] = sum * (1.0f / 1024.0f);
}

extern "C" void kernel_launch(void* const* d_in, const int* in_sizes, int n_in,
                              void* d_out, int out_size, void* d_ws, size_t ws_size,
                              hipStream_t stream) {
    (void)in_sizes; (void)n_in; (void)out_size; (void)ws_size;
    const float* x        = (const float*)d_in[0];
    const float* Wqkv     = (const float*)d_in[1];
    const float* Wproj    = (const float*)d_in[2];
    const float* cascadeW = (const float*)d_in[3];
    float* out = (float*)d_out;

    char* ws = (char*)d_ws;
    size_t off = 0;
    auto alloc = [&](size_t bytes) {
        void* p = ws + off;
        off = (off + bytes + 255) & ~size_t(255);
        return p;
    };
    _Float16* x16       = (_Float16*)alloc(4096UL * 1024 * 2);
    _Float16* WqkvT     = (_Float16*)alloc(3072UL * 1024 * 2);
    _Float16* WprojT    = (_Float16*)alloc(1024UL * 1024 * 2);
    float*    qkv32     = (float*)   alloc(4096UL * 3072 * 4);
    _Float16* q_rot     = (_Float16*)alloc(64UL * 1024 * 64 * 2);
    _Float16* k_rot     = (_Float16*)alloc(64UL * 1024 * 64 * 2);
    _Float16* v_t       = (_Float16*)alloc(64UL * 64 * 1024 * 2);
    float*    bias_part = (float*)   alloc(4UL * 64 * 1024 * 4);
    float*    bias      = (float*)   alloc(4UL * 1024 * 4);
    _Float16* attout    = x16;  // x16 dead after QKV GEMM; reuse for concat head outputs

    cast_f32_to_f16<<<(4096 * 1024) / 256, 256, 0, stream>>>(x, x16, 4096 * 1024);
    transpose_cast<<<(3072 * 1024) / 256, 256, 0, stream>>>(Wqkv, WqkvT, 1024, 3072);
    transpose_cast<<<(1024 * 1024) / 256, 256, 0, stream>>>(Wproj, WprojT, 1024, 1024);

    gemm_f16f32<<<dim3(3072 / 256, 4096 / 32), 256, 0, stream>>>(x16, WqkvT, qkv32, 4096, 3072, 1024);

    rope_split_kernel<<<(4 * 1024 * 16 * 32) / 256, 256, 0, stream>>>(qkv32, q_rot, k_rot, v_t);

    for (int h = 0; h < 16; h++) {
        if (h > 0) reduce_bias_kernel<<<16, 256, 0, stream>>>(bias_part, bias);
        attn_head_kernel<<<256, 256, 0, stream>>>(q_rot, k_rot, v_t, bias, cascadeW, attout, bias_part, h);
    }

    gemm_f16f32<<<dim3(1024 / 256, 4096 / 32), 256, 0, stream>>>(attout, WprojT, out, 4096, 1024, 1024);
}
